// Gaussian_RBF_62895501082691
// MI455X (gfx1250) — matrix operand order repaired and checked by reading, not compiled, not run on hardware
//
#include <hip/hip_runtime.h>
#include <hip/hip_bf16.h>

typedef __attribute__((ext_vector_type(16))) _Float16 v16h;
typedef __attribute__((ext_vector_type(8)))  float    v8f;

#define KDIM 256
#define TILE 128

// ---------------------------------------------------------------------------
// f32 -> hi/lo f16 split (row-major, for x / A-fragments)
// ---------------------------------------------------------------------------
__global__ __launch_bounds__(256) void split_f16_kernel(
    const float* __restrict__ in, _Float16* __restrict__ hi,
    _Float16* __restrict__ lo, int n) {
  int i = blockIdx.x * blockDim.x + threadIdx.x;
  if (i < n) {
    float v = in[i];
    _Float16 h = (_Float16)v;
    hi[i] = h;
    lo[i] = (_Float16)(v - (float)h);
  }
}

// ---------------------------------------------------------------------------
// f32 -> hi/lo f16 split + 32x32 tiled transpose (for y / B-fragments)
// out layout: [KDIM][rows]  (k-major), so B fragments are contiguous rows
// ---------------------------------------------------------------------------
__global__ __launch_bounds__(256) void split_T_f16_kernel(
    const float* __restrict__ in, _Float16* __restrict__ hiT,
    _Float16* __restrict__ loT, int rows) {
  __shared__ _Float16 Th[32][33];
  __shared__ _Float16 Tl[32][33];
  const int tx = threadIdx.x & 31;
  const int ty = threadIdx.x >> 5;          // 0..7
  const int nb = blockIdx.x * 32;           // row (n) tile base
  const int kb = blockIdx.y * 32;           // k tile base
#pragma unroll
  for (int r = 0; r < 4; ++r) {
    int n = nb + ty + 8 * r;
    float v = in[(size_t)n * KDIM + kb + tx];
    _Float16 h = (_Float16)v;
    Th[ty + 8 * r][tx] = h;
    Tl[ty + 8 * r][tx] = (_Float16)(v - (float)h);
  }
  __syncthreads();
#pragma unroll
  for (int r = 0; r < 4; ++r) {
    int k = kb + ty + 8 * r;
    hiT[(size_t)k * rows + nb + tx] = Th[tx][ty + 8 * r];
    loT[(size_t)k * rows + nb + tx] = Tl[tx][ty + 8 * r];
  }
}

// ---------------------------------------------------------------------------
// per-row squared L2 norm (f32 exact), one wave32 per row
// ---------------------------------------------------------------------------
__global__ __launch_bounds__(256) void rownorm_kernel(
    const float* __restrict__ in, float* __restrict__ out, int rows) {
  int w = (int)((blockIdx.x * blockDim.x + threadIdx.x) >> 5);
  int lane = threadIdx.x & 31;
  if (w >= rows) return;
  const float* p = in + (size_t)w * KDIM;
  float s = 0.f;
#pragma unroll
  for (int j = 0; j < KDIM / 32; ++j) {
    float v = p[lane + 32 * j];
    s = fmaf(v, v, s);
  }
#pragma unroll
  for (int off = 16; off > 0; off >>= 1) s += __shfl_xor(s, off);
  if (lane == 0) out[w] = s;
}

// ---------------------------------------------------------------------------
// main kernel: 128x128 tile of exp(-d2/2); WMMA hi/lo-split GEMM.
// Wave tiling 2x4 (2 row-tiles x 4 col-tiles per wave): each B fragment
// feeds 6 WMMAs -> 8 b128 A loads + 128 b16 B loads per 24 WMMAs per k-chunk.
// A fragments from row-major xh/xl; B fragments straight from k-major
// yhT/ylT (pre-transposed) -> zero LDS in the hot loop, L2/WGP$-resident.
// ---------------------------------------------------------------------------
__global__ __launch_bounds__(256) void rbf_wmma_kernel(
    const _Float16* __restrict__ xh, const _Float16* __restrict__ xl,
    const _Float16* __restrict__ yhT, const _Float16* __restrict__ ylT,
    const float* __restrict__ x2, const float* __restrict__ y2,
    float* __restrict__ partial, int Mtot) {
  __shared__ float redbuf[8];

  const int tid    = threadIdx.x;
  const int wave   = tid >> 5;
  const int wave_m = wave >> 1;        // 0..3 : row group (32 rows each)
  const int wave_n = wave & 1;         // 0..1 : col group (64 cols each)
  const int lane   = tid & 31;
  const int lhalf  = lane >> 4;        // 0: lanes 0-15, 1: lanes 16-31
  const int l16    = lane & 15;

  const int m_base = blockIdx.x * TILE;              // x rows
  const int n_base = blockIdx.y * TILE;              // y rows (output cols)
  const int m_wave = m_base + wave_m * 32;           // this wave's 32-row strip
  const int n_wave = n_base + wave_n * 64;           // this wave's 64-col strip

  v8f acc[2][4] = {};   // [row tile][col tile] 16x16 f32 accumulators

  // A-fragment rows (ISA: lanes<16 hold K 0-7 & 16-23 of chunk;
  // lanes>=16 hold K 8-15 & 24-31)
  const _Float16* pah0 = xh + (size_t)(m_wave + l16) * KDIM + 8 * lhalf;
  const _Float16* pal0 = xl + (size_t)(m_wave + l16) * KDIM + 8 * lhalf;
  const _Float16* pah1 = xh + (size_t)(m_wave + 16 + l16) * KDIM + 8 * lhalf;
  const _Float16* pal1 = xl + (size_t)(m_wave + 16 + l16) * KDIM + 8 * lhalf;

  union V16 { v16h v; uint4 q[2]; };

#pragma unroll
  for (int kc = 0; kc < KDIM; kc += 32) {
    // B-fragment layout: lane l holds column N = l16, K 0-7 & 16-23 (lanes<16)
    // or K 8-15 & 24-31 (lanes>=16) of the chunk, strided by Mtot in yT.
    const _Float16* pbh = yhT + (size_t)(kc + 8 * lhalf) * Mtot + n_wave + l16;
    const _Float16* pbl = ylT + (size_t)(kc + 8 * lhalf) * Mtot + n_wave + l16;

    if (kc + 32 < KDIM) {  // prefetch next chunk into WGP$ (global_prefetch_b8)
      __builtin_prefetch(pah0 + kc + 32, 0, 1);
      __builtin_prefetch(pah1 + kc + 32, 0, 1);
      __builtin_prefetch(yhT + (size_t)(kc + 32 + lane) * Mtot + n_wave, 0, 1);
      __builtin_prefetch(ylT + (size_t)(kc + 32 + lane) * Mtot + n_wave, 0, 1);
    }

    V16 a0h, a0l, a1h, a1l;
    a0h.q[0] = *(const uint4*)(pah0 + kc);
    a0h.q[1] = *(const uint4*)(pah0 + kc + 16);
    a0l.q[0] = *(const uint4*)(pal0 + kc);
    a0l.q[1] = *(const uint4*)(pal0 + kc + 16);
    a1h.q[0] = *(const uint4*)(pah1 + kc);
    a1h.q[1] = *(const uint4*)(pah1 + kc + 16);
    a1l.q[0] = *(const uint4*)(pal1 + kc);
    a1l.q[1] = *(const uint4*)(pal1 + kc + 16);

#pragma unroll
    for (int c = 0; c < 4; ++c) {
      V16 bh, bl;
#pragma unroll
      for (int s = 0; s < 16; ++s) {
        const int kk = (s < 8) ? s : s + 8;
        bh.v[s] = pbh[(size_t)kk * Mtot + c * 16];
        bl.v[s] = pbl[(size_t)kk * Mtot + c * 16];
      }
      // dot = hi*hi + hi*lo + lo*hi  (lo*lo ~ 2^-22, dropped)
      acc[0][c] = __builtin_amdgcn_wmma_f32_16x16x32_f16(
          false, a0h.v, false, bh.v, (short)0, acc[0][c], false, false);
      acc[0][c] = __builtin_amdgcn_wmma_f32_16x16x32_f16(
          false, a0h.v, false, bl.v, (short)0, acc[0][c], false, false);
      acc[0][c] = __builtin_amdgcn_wmma_f32_16x16x32_f16(
          false, a0l.v, false, bh.v, (short)0, acc[0][c], false, false);
      acc[1][c] = __builtin_amdgcn_wmma_f32_16x16x32_f16(
          false, a1h.v, false, bh.v, (short)0, acc[1][c], false, false);
      acc[1][c] = __builtin_amdgcn_wmma_f32_16x16x32_f16(
          false, a1h.v, false, bl.v, (short)0, acc[1][c], false, false);
      acc[1][c] = __builtin_amdgcn_wmma_f32_16x16x32_f16(
          false, a1l.v, false, bh.v, (short)0, acc[1][c], false, false);
    }
  }

  // epilogue: d2 = x2 + y2 - 2*dot, k = exp(-d2/2)
  // C layout: VGPR r -> M = r + 8*lhalf ; N = l16
  float x2v[2][8];
#pragma unroll
  for (int rt = 0; rt < 2; ++rt)
#pragma unroll
    for (int r = 0; r < 8; ++r)
      x2v[rt][r] = x2[m_wave + rt * 16 + 8 * lhalf + r];

  float local = 0.f;
#pragma unroll
  for (int c = 0; c < 4; ++c) {
    float y2v = y2[n_wave + c * 16 + l16];
#pragma unroll
    for (int rt = 0; rt < 2; ++rt) {
#pragma unroll
      for (int r = 0; r < 8; ++r) {
        float d2 = x2v[rt][r] + y2v - 2.0f * acc[rt][c][r];
        d2 = fmaxf(d2, 0.0f);
        local += __expf(-0.5f * d2);
      }
    }
  }

#pragma unroll
  for (int off = 16; off > 0; off >>= 1) local += __shfl_xor(local, off);
  if (lane == 0) redbuf[wave] = local;
  __syncthreads();
  if (tid == 0) {
    float s = 0.f;
#pragma unroll
    for (int w = 0; w < 8; ++w) s += redbuf[w];
    partial[blockIdx.y * gridDim.x + blockIdx.x] = s;
  }
}

// ---------------------------------------------------------------------------
// deterministic final reduction (no atomics -> graph-replay safe)
// ---------------------------------------------------------------------------
__global__ __launch_bounds__(256) void final_reduce_kernel(
    const float* __restrict__ partial, int n, float scale,
    float* __restrict__ out) {
  __shared__ float red[256];
  float s = 0.f;
  for (int i = threadIdx.x; i < n; i += 256) s += partial[i];
  red[threadIdx.x] = s;
  __syncthreads();
  for (int off = 128; off > 0; off >>= 1) {
    if ((int)threadIdx.x < off) red[threadIdx.x] += red[threadIdx.x + off];
    __syncthreads();
  }
  if (threadIdx.x == 0) out[0] = red[0] * scale;
}

// ---------------------------------------------------------------------------
extern "C" void kernel_launch(void* const* d_in, const int* in_sizes, int n_in,
                              void* d_out, int out_size, void* d_ws, size_t ws_size,
                              hipStream_t stream) {
  const float* x = (const float*)d_in[0];
  const float* y = (const float*)d_in[1];
  const int N = in_sizes[0] / KDIM;   // 8192
  const int M = in_sizes[1] / KDIM;   // 8192
  const size_t nx = (size_t)N * KDIM;
  const size_t ny = (size_t)M * KDIM;

  // workspace layout
  _Float16* xh  = (_Float16*)d_ws;
  _Float16* xl  = xh + nx;
  _Float16* yhT = xl + nx;             // [KDIM][M]
  _Float16* ylT = yhT + ny;            // [KDIM][M]
  float* x2      = (float*)(ylT + ny);
  float* y2      = x2 + N;
  float* partial = y2 + M;

  // 1) x: split f32 -> f16 hi/lo (row-major)
  split_f16_kernel<<<(int)((nx + 255) / 256), 256, 0, stream>>>(x, xh, xl, (int)nx);

  // 2) y: split + transpose to k-major
  {
    dim3 g(M / 32, KDIM / 32);
    split_T_f16_kernel<<<g, 256, 0, stream>>>(y, yhT, ylT, M);
  }

  // 3) exact f32 row norms
  rownorm_kernel<<<(N * 32 + 255) / 256, 256, 0, stream>>>(x, x2, N);
  rownorm_kernel<<<(M * 32 + 255) / 256, 256, 0, stream>>>(y, y2, M);

  // 4) WMMA RBF tiles
  dim3 grid(N / TILE, M / TILE);
  rbf_wmma_kernel<<<grid, 256, 0, stream>>>(xh, xl, yhT, ylT, x2, y2, partial, M);

  // 5) final mean
  const int nblocks = (N / TILE) * (M / TILE);
  const float scale = 1.0f / ((float)((long long)N * (long long)M));
  final_reduce_kernel<<<1, 256, 0, stream>>>(partial, nblocks, scale, (float*)d_out);
}
